// Compressor_77395310674149
// MI455X (gfx1250) — matrix operand order repaired and checked by reading, not compiled, not run on hardware
//
#include <hip/hip_runtime.h>
#include <hip/hip_bf16.h>
#include <stdint.h>

// ---------- problem constants ----------
#define T_TOK   8192
#define DIMK    7168
#define E_DIM   1024            // COFF * HEAD_DIM
#define N_CMP   2048
#define NSLOTS  65536

// ---------- workspace layout (bytes) ----------
#define OFF_XB    0ULL                     // 8192*7168 bf16 = 117,440,512
#define OFF_WKVB  117440512ULL             // 1024*7168 bf16 =  14,680,064
#define OFF_WGB   132120576ULL             // 1024*7168 bf16 =  14,680,064
#define OFF_APEB  146800640ULL             // 1024 f32
#define OFF_KV    146804736ULL             // 2048*1024 f32 = 8,388,608
// total ~155.2 MB

typedef __bf16 bf16_t;
typedef __attribute__((ext_vector_type(16))) __bf16 v16bf;
typedef __attribute__((ext_vector_type(8)))  float  v8f;

struct U2 { unsigned x, y; };
struct U4 { unsigned x, y, z, w; };
struct BF4 { bf16_t a, b, c, d; };

union Frag {
    v16bf    v;
    U2       u2[4];
    unsigned u[8];
};

// ============================================================
// 1) fp32 -> bf16 conversion for x, wkv, wgate (grid-stride, float4)
// ============================================================
__global__ void convert_f32_bf16(const float* __restrict__ x,
                                 const float* __restrict__ wkv,
                                 const float* __restrict__ wg,
                                 bf16_t* __restrict__ xb,
                                 bf16_t* __restrict__ wkvb,
                                 bf16_t* __restrict__ wgb) {
    const long NX4 = (long)T_TOK * DIMK / 4;
    const long NW4 = (long)E_DIM * DIMK / 4;
    const long TOT = NX4 + 2 * NW4;
    const long stride = (long)gridDim.x * blockDim.x;
    for (long i = (long)blockIdx.x * blockDim.x + threadIdx.x; i < TOT; i += stride) {
        const float4* src; bf16_t* dst; long j;
        if (i < NX4)            { src = (const float4*)x;   dst = xb;   j = i; }
        else if (i < NX4 + NW4) { src = (const float4*)wkv; dst = wkvb; j = i - NX4; }
        else                    { src = (const float4*)wg;  dst = wgb;  j = i - NX4 - NW4; }
        float4 v = src[j];
        BF4 o; o.a = (bf16_t)v.x; o.b = (bf16_t)v.y; o.c = (bf16_t)v.z; o.d = (bf16_t)v.w;
        *(BF4*)(dst + 4 * j) = o;
    }
}

// ============================================================
// 2) apebar[e] = mean_r ape[r][e]   (mean over R commutes with +ape)
// ============================================================
__global__ void ape_mean(const float* __restrict__ ape, float* __restrict__ apebar) {
    int e = blockIdx.x * blockDim.x + threadIdx.x;
    if (e < E_DIM)
        apebar[e] = 0.25f * (ape[e] + ape[E_DIM + e] + ape[2 * E_DIM + e] + ape[3 * E_DIM + e]);
}

// ============================================================
// 3) fused dual-GEMM (proj & gate) + sigmoid gating + R=4 mean + apebar
//    block = 256 threads (8 waves), tile = 256(M) x 64(N), K-step 32.
//    Each wave owns TWO 16-row C blocks -> 16 WMMAs per K-step
//    (5 LDS dwords per WMMA). Double-buffered LDS, one barrier per K-step.
// ============================================================
__launch_bounds__(256)
__global__ void gemm_gate_mean(const bf16_t* __restrict__ xb,
                               const bf16_t* __restrict__ wkvb,
                               const bf16_t* __restrict__ wgb,
                               const float*  __restrict__ apebar,
                               float*        __restrict__ kv) {
    // LDS tiles (double buffered), rows padded to 18 dwords (data = 16 dwords)
    __shared__ unsigned sA[2][256 * 18];   // 36,864 B
    __shared__ unsigned sK[2][64 * 18];    //  9,216 B
    __shared__ unsigned sG[2][64 * 18];    //  9,216 B

    const int m0 = blockIdx.y * 256;
    const int n0 = blockIdx.x * 64;
    const int t  = threadIdx.x;
    const int lane = t & 31;
    const int wv   = t >> 5;      // wave id -> 32-row block
    const int ln   = lane & 15;
    const int lh   = lane >> 4;   // lane half selects K interleave

    // cooperative-load assignments
    const int ra = t >> 1, ha = t & 1;  // A: rows ra and ra+128, 16 bf16 halves
    const int rb = t >> 2, qb = t & 3;  // B: 64 rows, 4 threads/row, 8 bf16 each

    const bf16_t* aptr0 = xb + (long)(m0 + ra) * DIMK + ha * 16;
    const bf16_t* aptr1 = aptr0 + (long)128 * DIMK;
    const bf16_t* kptr  = wkvb + (long)(n0 + rb) * DIMK + qb * 8;
    const bf16_t* gptr  = wgb  + (long)(n0 + rb) * DIMK + qb * 8;

    v8f pacc[2][4] = {};
    v8f gacc[2][4] = {};

    // ---- prologue: fetch tile 0 and park it in buffer 0 ----
    U4 pa0 = *(const U4*)(aptr0);
    U4 pa1 = *(const U4*)(aptr0 + 8);
    U4 pa2 = *(const U4*)(aptr1);
    U4 pa3 = *(const U4*)(aptr1 + 8);
    U4 pk  = *(const U4*)(kptr);
    U4 pg  = *(const U4*)(gptr);
    {
        unsigned* dA0 = &sA[0][ra * 18 + ha * 8];
        *(U2*)(dA0 + 0) = U2{pa0.x, pa0.y};
        *(U2*)(dA0 + 2) = U2{pa0.z, pa0.w};
        *(U2*)(dA0 + 4) = U2{pa1.x, pa1.y};
        *(U2*)(dA0 + 6) = U2{pa1.z, pa1.w};
        unsigned* dA1 = &sA[0][(ra + 128) * 18 + ha * 8];
        *(U2*)(dA1 + 0) = U2{pa2.x, pa2.y};
        *(U2*)(dA1 + 2) = U2{pa2.z, pa2.w};
        *(U2*)(dA1 + 4) = U2{pa3.x, pa3.y};
        *(U2*)(dA1 + 6) = U2{pa3.z, pa3.w};
        unsigned* dK = &sK[0][rb * 18 + qb * 4];
        *(U2*)(dK + 0) = U2{pk.x, pk.y};
        *(U2*)(dK + 2) = U2{pk.z, pk.w};
        unsigned* dG = &sG[0][rb * 18 + qb * 4];
        *(U2*)(dG + 0) = U2{pg.x, pg.y};
        *(U2*)(dG + 2) = U2{pg.z, pg.w};
    }
    __syncthreads();

    int cur = 0;
    for (int k0 = 0; k0 < DIMK; k0 += 32) {
        const bool more = (k0 + 32 < DIMK);

        // 1) fragments for this K-step
        //    A: 16-bit 16x32 layout (lanes 0-15: K 0-7/16-23; lanes 16-31: 8-15/24-31)
        Frag a[2], bk4[4], bg4[4];
        #pragma unroll
        for (int m = 0; m < 2; ++m) {
            const unsigned* base = &sA[cur][(wv * 32 + m * 16 + ln) * 18 + lh * 4];
            a[m].u2[0] = *(const U2*)(base + 0);
            a[m].u2[1] = *(const U2*)(base + 2);
            a[m].u2[2] = *(const U2*)(base + 8);
            a[m].u2[3] = *(const U2*)(base + 10);
        }
        #pragma unroll
        for (int c = 0; c < 4; ++c) {
            const unsigned* p = &sK[cur][(c * 16 + ln) * 18 + lh * 4];
            bk4[c].u2[0] = *(const U2*)(p + 0);
            bk4[c].u2[1] = *(const U2*)(p + 2);
            bk4[c].u2[2] = *(const U2*)(p + 8);
            bk4[c].u2[3] = *(const U2*)(p + 10);
            const unsigned* q = &sG[cur][(c * 16 + ln) * 18 + lh * 4];
            bg4[c].u2[0] = *(const U2*)(q + 0);
            bg4[c].u2[1] = *(const U2*)(q + 2);
            bg4[c].u2[2] = *(const U2*)(q + 8);
            bg4[c].u2[3] = *(const U2*)(q + 10);
        }

        // 2) issue next tile's global loads (in flight during the WMMAs)
        if (more) {
            aptr0 += 32; aptr1 += 32; kptr += 32; gptr += 32;
            pa0 = *(const U4*)(aptr0);
            pa1 = *(const U4*)(aptr0 + 8);
            pa2 = *(const U4*)(aptr1);
            pa3 = *(const U4*)(aptr1 + 8);
            pk  = *(const U4*)(kptr);
            pg  = *(const U4*)(gptr);
        }

        // 3) 16 WMMAs; each B fragment feeds two (different A), so even a
        //    load->wait->wmma schedule amortizes LDS latency over 2 matrix ops
        #pragma unroll
        for (int c = 0; c < 4; ++c) {
            #pragma unroll
            for (int m = 0; m < 2; ++m)
                pacc[m][c] = __builtin_amdgcn_wmma_f32_16x16x32_bf16(
                    false, a[m].v, false, bk4[c].v, (short)0, pacc[m][c], false, false);
            #pragma unroll
            for (int m = 0; m < 2; ++m)
                gacc[m][c] = __builtin_amdgcn_wmma_f32_16x16x32_bf16(
                    false, a[m].v, false, bg4[c].v, (short)0, gacc[m][c], false, false);
        }

        // 4) park next tile in the other buffer
        if (more) {
            unsigned* dA0 = &sA[cur ^ 1][ra * 18 + ha * 8];
            *(U2*)(dA0 + 0) = U2{pa0.x, pa0.y};
            *(U2*)(dA0 + 2) = U2{pa0.z, pa0.w};
            *(U2*)(dA0 + 4) = U2{pa1.x, pa1.y};
            *(U2*)(dA0 + 6) = U2{pa1.z, pa1.w};
            unsigned* dA1 = &sA[cur ^ 1][(ra + 128) * 18 + ha * 8];
            *(U2*)(dA1 + 0) = U2{pa2.x, pa2.y};
            *(U2*)(dA1 + 2) = U2{pa2.z, pa2.w};
            *(U2*)(dA1 + 4) = U2{pa3.x, pa3.y};
            *(U2*)(dA1 + 6) = U2{pa3.z, pa3.w};
            unsigned* dK = &sK[cur ^ 1][rb * 18 + qb * 4];
            *(U2*)(dK + 0) = U2{pk.x, pk.y};
            *(U2*)(dK + 2) = U2{pk.z, pk.w};
            unsigned* dG = &sG[cur ^ 1][rb * 18 + qb * 4];
            *(U2*)(dG + 0) = U2{pg.x, pg.y};
            *(U2*)(dG + 2) = U2{pg.z, pg.w};
        }

        // 5) single barrier per K-step
        __syncthreads();
        cur ^= 1;
    }

    // epilogue: sigmoid gate, mean over R=4 rows (in-lane: a 4-row group sits
    // in 4 consecutive acc VGPRs of one lane), + apebar. 8 kv rows per wave.
    #pragma unroll
    for (int m = 0; m < 2; ++m) {
        const int kvrow0 = blockIdx.y * 64 + wv * 8 + m * 4 + lh * 2;
        #pragma unroll
        for (int c = 0; c < 4; ++c) {
            float sLo = 0.f, sHi = 0.f;
            #pragma unroll
            for (int r = 0; r < 4; ++r) {
                float g1 = pacc[m][c][r]     * (1.f / (1.f + __expf(-gacc[m][c][r])));
                float g2 = pacc[m][c][r + 4] * (1.f / (1.f + __expf(-gacc[m][c][r + 4])));
                sLo += g1; sHi += g2;
            }
            const int col = n0 + c * 16 + ln;
            const float ab = apebar[col];
            kv[(long)kvrow0 * E_DIM + col]       = 0.25f * sLo + ab;
            kv[(long)(kvrow0 + 1) * E_DIM + col] = 0.25f * sHi + ab;
        }
    }
}

// ============================================================
// 4) per-(token, head) RMS-norm + RoPE(last 64) + scatter into cache
//    block = 128 threads, one block per (n, c)
// ============================================================
__global__ void norm_rope_scatter(const float* __restrict__ kv,
                                  const float* __restrict__ nw,
                                  const float* __restrict__ rc,
                                  const float* __restrict__ rs,
                                  const int*   __restrict__ slots,
                                  float*       __restrict__ out) {
    __shared__ float y[512];
    __shared__ float red[128];
    const int n = blockIdx.x >> 1;
    const int c = blockIdx.x & 1;
    const int t = threadIdx.x;
    const int base = t * 4;

    const float4 v = *(const float4*)(kv + (long)n * E_DIM + c * 512 + base);
    y[base + 0] = v.x; y[base + 1] = v.y; y[base + 2] = v.z; y[base + 3] = v.w;
    red[t] = v.x * v.x + v.y * v.y + v.z * v.z + v.w * v.w;
    __syncthreads();
    #pragma unroll
    for (int off = 64; off > 0; off >>= 1) {
        if (t < off) red[t] += red[t + off];
        __syncthreads();
    }
    const float scale = rsqrtf(red[0] * (1.f / 512.f) + 1e-6f);

    // normalize in place (each thread touches only its own 4 slots)
    #pragma unroll
    for (int j = 0; j < 4; ++j) y[base + j] = y[base + j] * scale * nw[base + j];
    __syncthreads();

    float o[4];
    #pragma unroll
    for (int j = 0; j < 4; ++j) {
        const int h = base + j;
        if (h < 448) {
            o[j] = y[h];
        } else {
            const int rj = h - 448;                 // 0..63 within rope segment
            const float cs = rc[n * 64 + rj];
            const float sn = rs[n * 64 + rj];
            o[j] = (rj < 32) ? (y[h] * cs - y[h + 32] * sn)
                             : (y[h] * cs + y[h - 32] * sn);
        }
    }
    const long obase = (long)slots[n] * E_DIM + c * 512 + base;
    *(float4*)(out + obase) = make_float4(o[0], o[1], o[2], o[3]);
}

// ============================================================
// launcher
// ============================================================
extern "C" void kernel_launch(void* const* d_in, const int* in_sizes, int n_in,
                              void* d_out, int out_size, void* d_ws, size_t ws_size,
                              hipStream_t stream) {
    (void)in_sizes; (void)n_in; (void)out_size; (void)ws_size;
    const float* x     = (const float*)d_in[0];
    const float* wkv   = (const float*)d_in[1];
    const float* wgate = (const float*)d_in[2];
    const float* ape   = (const float*)d_in[3];
    const float* nw    = (const float*)d_in[4];
    const float* rc    = (const float*)d_in[5];
    const float* rs    = (const float*)d_in[6];
    const float* cache = (const float*)d_in[7];
    const int*   slots = (const int*)d_in[8];
    float* out = (float*)d_out;

    char* ws = (char*)d_ws;
    bf16_t* xb    = (bf16_t*)(ws + OFF_XB);
    bf16_t* wkvb  = (bf16_t*)(ws + OFF_WKVB);
    bf16_t* wgb   = (bf16_t*)(ws + OFF_WGB);
    float*  apeb  = (float*)(ws + OFF_APEB);
    float*  kvbuf = (float*)(ws + OFF_KV);

    // background: untouched cache rows (input is zeros)
    hipMemcpyAsync(out, cache, (size_t)NSLOTS * E_DIM * sizeof(float),
                   hipMemcpyDeviceToDevice, stream);

    convert_f32_bf16<<<2048, 256, 0, stream>>>(x, wkv, wgate, xb, wkvb, wgb);
    ape_mean<<<4, 256, 0, stream>>>(ape, apeb);

    dim3 grid(E_DIM / 64, T_TOK / 256);   // 16 x 32 blocks
    gemm_gate_mean<<<grid, 256, 0, stream>>>(xb, wkvb, wgb, apeb, kvbuf);

    norm_rope_scatter<<<N_CMP * 2, 128, 0, stream>>>(kvbuf, nw, rc, rs, slots, out);
}
